// A_softmax_MHE_ce_head_50568944943168
// MI455X (gfx1250) — compile-verified, matrix operand order checked by reading, not run on hardware
//
#include <hip/hip_runtime.h>
#include <hip/hip_bf16.h>
#include <math.h>

#define EMB 512
#define CLS 20000
#define BSZ 256
#define PI_F 3.14159265f
#define LAMB_F (1500.0f / 1.1f)
#define LMD_INTER 0.01f

typedef __bf16 bf16;
typedef __attribute__((ext_vector_type(16))) __bf16 v16bf;
typedef __attribute__((ext_vector_type(8)))  __bf16 v8bf;
typedef __attribute__((ext_vector_type(8)))  float  v8f;

// ---------------- helpers ----------------

__device__ __forceinline__ v16bf load_frag16(const bf16* __restrict__ p) {
    // 16 bf16 per lane: 8 at p, 8 at p+16 (ISA 16-bit A/B fragment layout:
    // lane-half selects K {0..7,16..23} vs {8..15,24..31})
    v8bf lo = *(const v8bf*)(p);
    v8bf hi = *(const v8bf*)(p + 16);
    return __builtin_shufflevector(lo, hi, 0,1,2,3,4,5,6,7,8,9,10,11,12,13,14,15);
}

// ---------------- kernel 0: zero accumulators ----------------
__global__ void k_init(float* __restrict__ accums) {
    if (threadIdx.x < 4) accums[threadIdx.x] = 0.0f;
}

// ---------------- kernel 1: inverse column norms of W ----------------
__global__ void k_wnorm(const float* __restrict__ W, float* __restrict__ invWnorm) {
    int c = blockIdx.x * blockDim.x + threadIdx.x;
    if (c >= CLS) return;
    float s = 0.0f;
    for (int e = 0; e < EMB; ++e) {
        float v = W[(size_t)e * CLS + c];   // coalesced across c
        s = fmaf(v, v, s);
    }
    invWnorm[c] = rsqrtf(s);
}

// ---------------- kernel 2: emb row norms + bf16 convert ----------------
__global__ void k_xnorm(const float* __restrict__ emb, float* __restrict__ xlen,
                        float* __restrict__ invxlen, bf16* __restrict__ embB) {
    __shared__ float red[256];
    int r = blockIdx.x;
    int tid = threadIdx.x;
    const float* row = emb + (size_t)r * EMB;
    float s = 0.0f;
    for (int e = tid; e < EMB; e += 256) {
        float v = row[e];
        s = fmaf(v, v, s);
        embB[(size_t)r * EMB + e] = (bf16)v;
    }
    red[tid] = s;
    __syncthreads();
    for (int st = 128; st > 0; st >>= 1) {
        if (tid < st) red[tid] += red[tid + st];
        __syncthreads();
    }
    if (tid == 0) {
        float n = sqrtf(red[0]);
        xlen[r] = n;
        invxlen[r] = 1.0f / n;
    }
}

// ---------------- kernel 3: transpose + normalize W -> bf16 K-major ----------------
// Wn[c*EMB + e] = bf16(W[e*CLS + c] * invWnorm[c])
__global__ void k_normW(const float* __restrict__ W, const float* __restrict__ invWnorm,
                        bf16* __restrict__ Wn) {
    __shared__ float tile[32][33];
    int c0 = blockIdx.x * 32;
    int e0 = blockIdx.y * 32;
    int tx = threadIdx.x;   // 0..31
    int ty = threadIdx.y;   // 0..7
    #pragma unroll
    for (int i = 0; i < 4; ++i) {
        int e = e0 + ty + 8 * i;
        tile[ty + 8 * i][tx] = W[(size_t)e * CLS + (c0 + tx)];   // coalesced read
    }
    __syncthreads();
    #pragma unroll
    for (int i = 0; i < 4; ++i) {
        int c = c0 + ty + 8 * i;
        float iw = invWnorm[c];
        Wn[(size_t)c * EMB + (e0 + tx)] = (bf16)(tile[tx][ty + 8 * i] * iw);  // coalesced write
    }
}

// ---------------- kernel 4: gather A_sel = Wn[y[i]] ----------------
__global__ void k_gather(const bf16* __restrict__ Wn, const int* __restrict__ y,
                         bf16* __restrict__ Asel) {
    int i = blockIdx.x;
    int t = y[i];
    const bf16* src = Wn + (size_t)t * EMB;
    bf16* dst = Asel + (size_t)i * EMB;
    for (int e = threadIdx.x; e < EMB; e += 256) dst[e] = src[e];
}

// ---------------- WMMA GEMM (MODE 0: cos_s logits, MODE 1: MHE inter sum) ----------
// Block: 256 threads = 8 waves. Block tile 16(M) x 512(N); wave tile 16 x 64.
// A tile (16 x 512 bf16, 16 KB) staged in LDS once. k-loop FULLY UNROLLED
// (16 stages) with an explicit 2-deep B pipeline so load destinations feed
// WMMA operands directly (no loop-carried phi copies). Out-of-range column
// tiles are CLAMPED so the whole pipeline is branch-free; only the epilogue
// checks validity.
template <int MODE>
__global__ void k_gemm(const bf16* __restrict__ A, const bf16* __restrict__ Bm,
                       const float* __restrict__ xlen, const float* __restrict__ invxlen,
                       const int* __restrict__ y,
                       float* __restrict__ outLogits, float* __restrict__ accums) {
    __shared__ bf16 sA[16 * EMB];   // 16 KB

    const int tid  = threadIdx.x;
    const int lane = tid & 31;
    const int wave = tid >> 5;
    const int m0 = blockIdx.y * 16;
    const int n0 = blockIdx.x * 512 + wave * 64;

    // cooperative A-tile stage: 1024 x 16B
    {
        const uint4* gA = (const uint4*)(A + (size_t)m0 * EMB);
        uint4* s4 = (uint4*)sA;
        #pragma unroll
        for (int i = 0; i < 4; ++i) s4[tid + 256 * i] = gA[tid + 256 * i];
    }
    __syncthreads();

    const int nlane = lane & 15;
    const int hi    = lane >> 4;
    const int khalf = hi * 8;

    bool stv[4];
    const bf16* bp[4];
    #pragma unroll
    for (int j = 0; j < 4; ++j) {
        int nb = n0 + 16 * j;
        stv[j] = nb < CLS;                       // CLS % 16 == 0
        int nbc = stv[j] ? nb : (CLS - 16);      // clamp: harmless duplicate work
        bp[j] = Bm + (size_t)(nbc + nlane) * EMB + khalf;
    }
    const bf16* aBase = sA + (size_t)nlane * EMB + khalf;

    v8f acc[4] = {};
    v16bf buf[2][4];

    // prologue: stage 0 B fragments
    #pragma unroll
    for (int j = 0; j < 4; ++j) buf[0][j] = load_frag16(bp[j]);

    // 16 fully unrolled K stages, software-pipelined one stage ahead
    #pragma unroll
    for (int s = 0; s < EMB / 32; ++s) {
        const int cur = s & 1;
        const int nxt = cur ^ 1;
        if (s + 1 < EMB / 32) {
            #pragma unroll
            for (int j = 0; j < 4; ++j)
                buf[nxt][j] = load_frag16(bp[j] + 32 * (s + 1));
        }
        v16bf afrag = load_frag16(aBase + 32 * s);
        #pragma unroll
        for (int j = 0; j < 4; ++j)
            acc[j] = __builtin_amdgcn_wmma_f32_16x16x32_bf16(
                false, afrag, false, buf[cur][j], (short)0, acc[j], false, false);
    }

    if (MODE == 0) {
        // cos_s = clip(dot / xlen, -1, 1) * xlen   (Wn already column-normalized)
        #pragma unroll
        for (int j = 0; j < 4; ++j) {
            if (!stv[j]) continue;
            int col = n0 + 16 * j + nlane;
            #pragma unroll
            for (int r = 0; r < 8; ++r) {
                int row = m0 + 8 * hi + r;
                float cosv = fminf(fmaxf(acc[j][r] * invxlen[row], -1.0f), 1.0f);
                outLogits[(size_t)row * CLS + col] = cosv * xlen[row];
            }
        }
    } else {
        // inter: sum over (i, c != y[i]) of 1 / max(2 - 2*cosWW, 0)
        int yv[8];
        #pragma unroll
        for (int r = 0; r < 8; ++r) yv[r] = y[m0 + 8 * hi + r];
        float local = 0.0f;
        #pragma unroll
        for (int j = 0; j < 4; ++j) {
            if (!stv[j]) continue;
            int col = n0 + 16 * j + nlane;
            #pragma unroll
            for (int r = 0; r < 8; ++r) {
                float dist2 = fmaxf(2.0f - 2.0f * acc[j][r], 0.0f);
                float inv = (col == yv[r]) ? 0.0f : (1.0f / dist2);
                local += inv;
            }
        }
        #pragma unroll
        for (int s = 16; s > 0; s >>= 1) local += __shfl_xor(local, s, 32);
        if (lane == 0) atomicAdd(&accums[2], local);
    }
}

// ---------------- kernel 7: per-row margin + log-softmax + argmax ----------------
__global__ void k_softmax_ce(const float* __restrict__ logits, const int* __restrict__ y,
                             const float* __restrict__ xlen, const float* __restrict__ invxlen,
                             float* __restrict__ accums) {
    __shared__ float red_f[256];
    __shared__ int   red_i[256];
    __shared__ float s_outt;
    __shared__ int   s_pred;

    int r = blockIdx.x;
    int tid = threadIdx.x;
    int t = y[r];
    const float* row = logits + (size_t)r * CLS;
    float xl = xlen[r], ix = invxlen[r];

    if (tid == 0) {
        float cs = row[t];
        float c = fminf(fmaxf(cs * ix, -1.0f), 1.0f);
        float th = acosf(c);
        float k = floorf(4.0f * th / PI_F);
        float sign = 1.0f - 2.0f * fmodf(k, 2.0f);
        float c2 = c * c;
        float cosm = 8.0f * c2 * c2 - 8.0f * c2 + 1.0f;   // cos(4*theta)
        float phi = sign * cosm - 2.0f * k;
        float f = 1.0f / (1.0f + LAMB_F);
        s_outt = cs + f * (phi * xl - cs);
    }
    __syncthreads();
    float outt = s_outt;

    // pass 1: argmax of cos_s; max of margined output
    float bestv = -INFINITY; int besti = CLS;
    float mmax = -INFINITY;
    for (int c0 = tid; c0 < CLS; c0 += 256) {
        float v = row[c0];
        if (v > bestv || (v == bestv && c0 < besti)) { bestv = v; besti = c0; }
        float ov = (c0 == t) ? outt : v;
        mmax = fmaxf(mmax, ov);
    }
    red_f[tid] = bestv; red_i[tid] = besti;
    __syncthreads();
    for (int st = 128; st > 0; st >>= 1) {
        if (tid < st) {
            float v2 = red_f[tid + st]; int i2 = red_i[tid + st];
            if (v2 > red_f[tid] || (v2 == red_f[tid] && i2 < red_i[tid])) {
                red_f[tid] = v2; red_i[tid] = i2;
            }
        }
        __syncthreads();
    }
    if (tid == 0) s_pred = red_i[0];
    __syncthreads();

    red_f[tid] = mmax;
    __syncthreads();
    for (int st = 128; st > 0; st >>= 1) {
        if (tid < st) red_f[tid] = fmaxf(red_f[tid], red_f[tid + st]);
        __syncthreads();
    }
    float M = red_f[0];
    __syncthreads();

    // pass 2: sum exp
    float s = 0.0f;
    for (int c0 = tid; c0 < CLS; c0 += 256) {
        float v = row[c0];
        float ov = (c0 == t) ? outt : v;
        s += expf(ov - M);
    }
    red_f[tid] = s;
    __syncthreads();
    for (int st = 128; st > 0; st >>= 1) {
        if (tid < st) red_f[tid] += red_f[tid + st];
        __syncthreads();
    }
    if (tid == 0) {
        float logpt = (outt - M) - logf(red_f[0]);
        atomicAdd(&accums[0], -logpt);
        if (s_pred == t) atomicAdd(&accums[1], 1.0f);
    }
}

// ---------------- kernel 8: finalize scalars ----------------
__global__ void k_finalize(const float* __restrict__ accums, float* __restrict__ out) {
    float inter = accums[2] / (float)((long long)BSZ * (CLS - 1));
    out[0] = accums[0] / (float)BSZ + LMD_INTER * inter;            // loss
    out[1 + (size_t)BSZ * CLS] = accums[1] / (float)BSZ;            // acc
    out[2 + (size_t)BSZ * CLS] = inter;                             // inter
}

// ---------------- launcher ----------------
extern "C" void kernel_launch(void* const* d_in, const int* in_sizes, int n_in,
                              void* d_out, int out_size, void* d_ws, size_t ws_size,
                              hipStream_t stream) {
    const float* emb = (const float*)d_in[0];
    const int*   y   = (const int*)d_in[1];
    const float* W   = (const float*)d_in[2];
    float* out = (float*)d_out;
    float* logits = out + 1;   // [loss | logits(B*CLS) | acc | inter]

    char* ws = (char*)d_ws;
    float* invWnorm = (float*)(ws + 0);              //  80000 B
    float* xlen     = (float*)(ws + 80128);          //   1024 B
    float* invxlen  = (float*)(ws + 81152);          //   1024 B
    float* accums   = (float*)(ws + 82176);          //     16 B
    bf16*  embB     = (bf16*) (ws + 82432);          // 262144 B
    bf16*  Asel     = (bf16*) (ws + 344576);         // 262144 B
    bf16*  Wn       = (bf16*) (ws + 606720);         // 20480000 B + 4096 pad (~20.1 MiB total)

    k_init<<<1, 32, 0, stream>>>(accums);
    k_wnorm<<<(CLS + 255) / 256, 256, 0, stream>>>(W, invWnorm);
    k_xnorm<<<BSZ, 256, 0, stream>>>(emb, xlen, invxlen, embB);
    k_normW<<<dim3(CLS / 32, EMB / 32), dim3(32, 8), 0, stream>>>(W, invWnorm, Wn);
    k_gather<<<BSZ, 256, 0, stream>>>(Wn, y, Asel);

    dim3 ggrid((CLS + 511) / 512, BSZ / 16);   // 40 x 16 blocks, 8 waves each
    k_gemm<0><<<ggrid, 256, 0, stream>>>(embB, Wn, xlen, invxlen, y, logits, accums);
    k_gemm<1><<<ggrid, 256, 0, stream>>>(Asel, Wn, xlen, invxlen, y, logits, accums);

    k_softmax_ce<<<BSZ, 256, 0, stream>>>(logits, y, xlen, invxlen, accums);
    k_finalize<<<1, 1, 0, stream>>>(accums, out);
}
